// TransformerBlock_79302276153475
// MI455X (gfx1250) — compile-verified, matrix operand order checked by reading, not run on hardware
//
#include <hip/hip_runtime.h>
#include <hip/hip_bf16.h>

#define B_ 2
#define N_ 8192
#define D_ 64
#define K_ 16

typedef __attribute__((ext_vector_type(16))) _Float16 v16h;
typedef __attribute__((ext_vector_type(8)))  float    v8f;

__device__ __forceinline__ v8f wmma_f16(v16h a, v16h b, v8f c) {
    // 8 args: (neg_a, A, neg_b, B, c_mod, C, reuse_a, reuse_b)
    return __builtin_amdgcn_wmma_f32_16x16x32_f16(false, a, false, b, (short)0, c, false, false);
}

// ---- async global->LDS copy (16B per lane), tracked by ASYNCcnt ----
__device__ __forceinline__ void async_copy16(void* lds_dst, const void* gsrc) {
    uint32_t l = (uint32_t)(uintptr_t)lds_dst;            // LDS byte offset
    uint64_t g = (uint64_t)(uintptr_t)gsrc;
    asm volatile("global_load_async_to_lds_b128 %0, %1, off"
                 :: "v"(l), "v"(g) : "memory");
}
__device__ __forceinline__ void async_wait0() {
    asm volatile("s_wait_asynccnt 0x0" ::: "memory");
}

struct f8 { float v[8]; };

// 8 contiguous floats via two b128 loads (p must be 16B aligned)
__device__ __forceinline__ f8 ld8(const float* p) {
    f8 r;
    const float4 a = *(const float4*)p;
    const float4 b = *(const float4*)(p + 4);
    r.v[0] = a.x; r.v[1] = a.y; r.v[2] = a.z; r.v[3] = a.w;
    r.v[4] = b.x; r.v[5] = b.y; r.v[6] = b.z; r.v[7] = b.w;
    return r;
}

// Build A-matrix (16x32 f16) fragment from f32 row slice.
// p points at row_base + ks + kb; elements = {p[0..7], p[16..23]} (b128 loads + cvt).
__device__ __forceinline__ v16h cvt_a(const float* p) {
    const f8 lo = ld8(p);
    const f8 hi = ld8(p + 16);
    v16h a;
#pragma unroll
    for (int e = 0; e < 8; ++e) {
        a[e]     = (_Float16)lo.v[e];
        a[e + 8] = (_Float16)hi.v[e];
    }
    return a;
}

// ---------------------------------------------------------------------------
// Weight pre-swizzle: pack every weight into per-lane WMMA B-fragment order.
// Global fragment id f = wsel*8 + ctile*2 + kchunk; per frag: 32 lanes x 16 f16.
// wsel: 0=delta_w1 (K rows 0..2, rest zero), 1=delta_w2, 2=gamma_w1, 3=gamma_w2,
//       4=w_qs, 5=w_ks, 6=w_vs
// ---------------------------------------------------------------------------
__global__ __launch_bounds__(32) void swz_weights_kernel(
        const float* __restrict__ dw1, const float* __restrict__ dw2,
        const float* __restrict__ gw1, const float* __restrict__ gw2,
        const float* __restrict__ wq,  const float* __restrict__ wk,
        const float* __restrict__ wv,  _Float16* __restrict__ swz) {
    const int f    = blockIdx.x;          // 0..55
    const int wsel = f >> 3;
    const int t    = (f >> 1) & 3;
    const int kc   = f & 1;
    const int lane = threadIdx.x;

    const float* W = (wsel == 0) ? dw1 : (wsel == 1) ? dw2 : (wsel == 2) ? gw1
                   : (wsel == 3) ? gw2 : (wsel == 4) ? wq  : (wsel == 5) ? wk : wv;
    const int krows = (wsel == 0) ? 3 : 64;

    const int col = t * 16 + (lane & 15);
    const int kb  = (lane < 16) ? 0 : 8;
    const int ks  = kc * 32;

    v16h b;
#pragma unroll
    for (int e = 0; e < 8; ++e) {
        const int k0 = ks + kb + e;
        const int k1 = k0 + 16;
        b[e]     = (k0 < krows) ? (_Float16)W[k0 * 64 + col] : (_Float16)0.f;
        b[e + 8] = (k1 < krows) ? (_Float16)W[k1 * 64 + col] : (_Float16)0.f;
    }
    *(v16h*)&swz[((size_t)f * 32 + lane) * 16] = b;
}

// ---------------------------------------------------------------------------
// KNN: one thread per query point; candidate tiles staged to LDS with
// GLOBAL_LOAD_ASYNC_TO_LDS_B128 (tile = contiguous 1536B); top-16 in registers.
// ---------------------------------------------------------------------------
#define KT 128
__global__ __launch_bounds__(KT) void knn_kernel(const float* __restrict__ xyz,
                                                 int* __restrict__ knn_idx) {
    __shared__ float sP[KT * 3];
    const int b = blockIdx.y;
    const int i = blockIdx.x * KT + threadIdx.x;
    const float* xb = xyz + (size_t)b * N_ * 3;
    const float qx = xb[i * 3 + 0], qy = xb[i * 3 + 1], qz = xb[i * 3 + 2];

    float bd[K_];
    int   bi[K_];
#pragma unroll
    for (int s = 0; s < K_; ++s) { bd[s] = 3.4e38f; bi[s] = 0; }

    for (int jt = 0; jt < N_; jt += KT) {
        __syncthreads();
        if (threadIdx.x < (KT * 3) / 4)   // 96 x 16B = one candidate tile
            async_copy16(((uint4*)sP) + threadIdx.x,
                         ((const uint4*)(xb + (size_t)jt * 3)) + threadIdx.x);
        async_wait0();
        __syncthreads();
        if (jt + KT < N_)  // hint next tile into cache (global_prefetch_b8)
            __builtin_prefetch(xb + (size_t)(jt + KT) * 3 + threadIdx.x * 3, 0, 0);
        for (int jj = 0; jj < KT; ++jj) {           // uniform -> LDS broadcast reads
            const float dx = qx - sP[jj * 3 + 0];
            const float dy = qy - sP[jj * 3 + 1];
            const float dz = qz - sP[jj * 3 + 2];
            float d = dx * dx + dy * dy + dz * dz;
            if (d < bd[K_ - 1]) {
                int cand = jt + jj;
#pragma unroll
                for (int s = 0; s < K_; ++s) {
                    if (d < bd[s]) {
                        float td = bd[s]; int ti = bi[s];
                        bd[s] = d; bi[s] = cand; d = td; cand = ti;
                    }
                }
            }
        }
    }
    int* out = knn_idx + ((size_t)b * N_ + i) * K_;
#pragma unroll
    for (int s = 0; s < K_; ++s) out[s] = bi[s];
}

// ---------------------------------------------------------------------------
// QKV projections: [16384,64]@[64,64] via WMMA; one wave computes 16 rows x 64
// B fragments async-copied to LDS once per block -> single v16h ds reads.
// ---------------------------------------------------------------------------
__global__ __launch_bounds__(128) void qkv_kernel(const float* __restrict__ feats,
                                                  const _Float16* __restrict__ swz,
                                                  float* __restrict__ q,
                                                  float* __restrict__ kf,
                                                  float* __restrict__ vf) {
    __shared__ v16h sB[8 * 32];           // 8 fragments of the selected weight (8 KB)
    float* Osel = (blockIdx.y == 0) ? q : (blockIdx.y == 1) ? kf : vf;
    {   // async copy fragments of weight wsel = 4 + blockIdx.y
        const uint4* src = (const uint4*)(swz + (size_t)(4 + blockIdx.y) * 8 * 32 * 16);
        uint4* dst = (uint4*)sB;
#pragma unroll
        for (int i = threadIdx.x; i < 8 * 32 * 2; i += 128)
            async_copy16(dst + i, src + i);
        async_wait0();
    }
    __syncthreads();

    const int lane = threadIdx.x & 31;
    const int w    = threadIdx.x >> 5;
    const int rowbase = (blockIdx.x * 4 + w) * 16;
    const int row     = rowbase + (lane & 15);
    const int kb      = (lane < 16) ? 0 : 8;

    v8f acc[4];
#pragma unroll
    for (int t = 0; t < 4; ++t)
#pragma unroll
        for (int r = 0; r < 8; ++r) acc[t][r] = 0.f;

#pragma unroll
    for (int kc = 0; kc < 2; ++kc) {
        v16h a = cvt_a(feats + (size_t)row * 64 + kc * 32 + kb);
#pragma unroll
        for (int t = 0; t < 4; ++t) {
            v16h bm = sB[(t * 2 + kc) * 32 + lane];
            acc[t] = wmma_f16(a, bm, acc[t]);
        }
    }
    const int roff = (lane < 16) ? 0 : 8;
#pragma unroll
    for (int t = 0; t < 4; ++t) {
        const int col = t * 16 + (lane & 15);
#pragma unroll
        for (int r = 0; r < 8; ++r)
            Osel[(size_t)(rowbase + roff + r) * 64 + col] = acc[t][r];
    }
}

// ---------------------------------------------------------------------------
// Fused point-transformer attention: one wave per point.
//   pos = MLP(xyz_q - xyz_nb)          [16x3]->[16x64]->[16x64]   (WMMA)
//   attn = softmax_k MLP(q - k + pos)  [16x64]->[16x64]->[16x64]  (WMMA)
//   out  = sum_k attn*(v+pos) + feats
// ---------------------------------------------------------------------------
#define WV 4
__global__ __launch_bounds__(128) void attn_kernel(
        const float* __restrict__ xyz, const float* __restrict__ feats,
        const int* __restrict__ knn_idx, const _Float16* __restrict__ swz,
        const float* __restrict__ q, const float* __restrict__ kf, const float* __restrict__ vf,
        const float* __restrict__ db1, const float* __restrict__ db2,
        const float* __restrict__ gb1, const float* __restrict__ gb2,
        float* __restrict__ out) {
    __shared__ v16h  sB[32 * 32];         // frags of dw1,dw2,gw1,gw2 (32 KB)
    __shared__ float sH[WV][16][68];      // per-wave layout-transpose buffer (+pad)
    __shared__ int   sIdx[WV][16];

    {   // async bulk copy of pre-swizzled fragments 0..31
        const uint4* src = (const uint4*)swz;
        uint4* dst = (uint4*)sB;
#pragma unroll
        for (int i = threadIdx.x; i < 32 * 32 * 2; i += 128)
            async_copy16(dst + i, src + i);
        async_wait0();
    }

    const int lane = threadIdx.x & 31;
    const int w    = threadIdx.x >> 5;
    const int p    = blockIdx.x * WV + w;       // flat point id = b*N + n
    const int b    = p >> 13;                   // N = 2^13
    const size_t prow = (size_t)p;

    if (lane < 16) sIdx[w][lane] = knn_idx[prow * K_ + lane];
    __syncthreads();

    const int m    = lane & 15;                 // neighbor row this lane serves in A-layout
    const int roff = (lane < 16) ? 0 : 8;       // C-layout row offset of this half-wave
    const int kb   = roff;                      // A/B-layout K band offset
    const int idxm = sIdx[w][m];

    // ---- pos MLP layer 1 (K=3 zero-padded to 32; padding baked into fragments) ----
    const float qx = xyz[prow * 3 + 0], qy = xyz[prow * 3 + 1], qz = xyz[prow * 3 + 2];
    const size_t nb = ((size_t)b * N_ + idxm) * 3;
    const float dx = qx - xyz[nb + 0], dy = qy - xyz[nb + 1], dz = qz - xyz[nb + 2];

    v16h a1;
#pragma unroll
    for (int e = 0; e < 16; ++e) a1[e] = (_Float16)0.f;
    if (lane < 16) { a1[0] = (_Float16)dx; a1[1] = (_Float16)dy; a1[2] = (_Float16)dz; }

    v8f acc[4];
#pragma unroll
    for (int t = 0; t < 4; ++t) {
        const float bias = db1[t * 16 + m];
#pragma unroll
        for (int r = 0; r < 8; ++r) acc[t][r] = bias;
        v16h bm = sB[(0 * 8 + t * 2 + 0) * 32 + lane];   // dw1, kc=0
        acc[t] = wmma_f16(a1, bm, acc[t]);
#pragma unroll
        for (int r = 0; r < 8; ++r) acc[t][r] = fmaxf(acc[t][r], 0.f);  // relu
    }
    // C-layout -> LDS [row][col]
#pragma unroll
    for (int t = 0; t < 4; ++t) {
        const int col = t * 16 + m;
#pragma unroll
        for (int r = 0; r < 8; ++r) sH[w][roff + r][col] = acc[t][r];
    }
    __syncthreads();

    // ---- pos MLP layer 2: H1 @ dw2 + db2 -> posC (kept in registers) ----
    v8f posC[4];
#pragma unroll
    for (int t = 0; t < 4; ++t) {
        const float bias = db2[t * 16 + m];
#pragma unroll
        for (int r = 0; r < 8; ++r) posC[t][r] = bias;
    }
#pragma unroll
    for (int kc = 0; kc < 2; ++kc) {
        v16h a = cvt_a(&sH[w][m][kc * 32 + kb]);
#pragma unroll
        for (int t = 0; t < 4; ++t) {
            v16h bm = sB[(1 * 8 + t * 2 + kc) * 32 + lane];  // dw2
            posC[t] = wmma_f16(a, bm, posC[t]);
        }
    }
    __syncthreads();
    // pos -> LDS so next layer can read it in A-layout
#pragma unroll
    for (int t = 0; t < 4; ++t) {
        const int col = t * 16 + m;
#pragma unroll
        for (int r = 0; r < 8; ++r) sH[w][roff + r][col] = posC[t][r];
    }
    __syncthreads();

    // ---- gamma layer 1: relu((q - k_gather + pos) @ gw1 + gb1) ----
    const float* qrow = q  + prow * 64;
    const float* krow = kf + ((size_t)b * N_ + idxm) * 64;
#pragma unroll
    for (int t = 0; t < 4; ++t) {
        const float bias = gb1[t * 16 + m];
#pragma unroll
        for (int r = 0; r < 8; ++r) acc[t][r] = bias;
    }
#pragma unroll
    for (int kc = 0; kc < 2; ++kc) {
        const int k0 = kc * 32 + kb;
        const f8 qlo = ld8(qrow + k0),      qhi = ld8(qrow + k0 + 16);
        const f8 klo = ld8(krow + k0),      khi = ld8(krow + k0 + 16);
        const f8 hlo = ld8(&sH[w][m][k0]),  hhi = ld8(&sH[w][m][k0 + 16]);
        v16h a;
#pragma unroll
        for (int e = 0; e < 8; ++e) {
            a[e]     = (_Float16)(qlo.v[e] - klo.v[e] + hlo.v[e]);
            a[e + 8] = (_Float16)(qhi.v[e] - khi.v[e] + hhi.v[e]);
        }
#pragma unroll
        for (int t = 0; t < 4; ++t) {
            v16h bm = sB[(2 * 8 + t * 2 + kc) * 32 + lane];  // gw1
            acc[t] = wmma_f16(a, bm, acc[t]);
        }
    }
    __syncthreads();
#pragma unroll
    for (int t = 0; t < 4; ++t) {
        const int col = t * 16 + m;
#pragma unroll
        for (int r = 0; r < 8; ++r) sH[w][roff + r][col] = fmaxf(acc[t][r], 0.f);
    }
    __syncthreads();

    // ---- gamma layer 2: attn logits ----
#pragma unroll
    for (int t = 0; t < 4; ++t) {
        const float bias = gb2[t * 16 + m];
#pragma unroll
        for (int r = 0; r < 8; ++r) acc[t][r] = bias;
    }
#pragma unroll
    for (int kc = 0; kc < 2; ++kc) {
        v16h a = cvt_a(&sH[w][m][kc * 32 + kb]);
#pragma unroll
        for (int t = 0; t < 4; ++t) {
            v16h bm = sB[(3 * 8 + t * 2 + kc) * 32 + lane];  // gw2
            acc[t] = wmma_f16(a, bm, acc[t]);
        }
    }

    // ---- softmax over the 16 neighbors (8 VGPRs x 2 half-waves) + weighted sum ----
#pragma unroll
    for (int t = 0; t < 4; ++t) {
        float mx = acc[t][0];
#pragma unroll
        for (int r = 1; r < 8; ++r) mx = fmaxf(mx, acc[t][r]);
        mx = fmaxf(mx, __shfl_xor(mx, 16, 32));
        float ev[8];
        float s = 0.f;
#pragma unroll
        for (int r = 0; r < 8; ++r) { ev[r] = __expf(acc[t][r] - mx); s += ev[r]; }
        s += __shfl_xor(s, 16, 32);
        const float inv = 1.0f / s;

        const int col = t * 16 + m;
        float rsum = 0.f;
#pragma unroll
        for (int r = 0; r < 8; ++r) {
            const int nbr = sIdx[w][roff + r];
            const float vv = vf[((size_t)b * N_ + nbr) * 64 + col];
            rsum += (ev[r] * inv) * (vv + posC[t][r]);
        }
        rsum += __shfl_xor(rsum, 16, 32);
        if (lane < 16) out[prow * 64 + col] = rsum + feats[prow * 64 + col];
    }
}

// ---------------------------------------------------------------------------
// BatchNorm over (B,N) per channel: stats pass + in-place apply
// ---------------------------------------------------------------------------
__global__ __launch_bounds__(256) void bn_stats_kernel(const float* __restrict__ x,
                                                       float* __restrict__ stats) {
    __shared__ float ss[256], s2[256];
    const int ch = blockIdx.x;
    float a = 0.f, b = 0.f;
    for (int r = threadIdx.x; r < B_ * N_; r += 256) {
        const float v = x[(size_t)r * 64 + ch];
        a += v; b += v * v;
    }
    ss[threadIdx.x] = a; s2[threadIdx.x] = b;
    __syncthreads();
    for (int st = 128; st > 0; st >>= 1) {
        if (threadIdx.x < st) {
            ss[threadIdx.x] += ss[threadIdx.x + st];
            s2[threadIdx.x] += s2[threadIdx.x + st];
        }
        __syncthreads();
    }
    if (threadIdx.x == 0) {
        const float inv_n = 1.0f / (float)(B_ * N_);
        const float mean = ss[0] * inv_n;
        stats[ch]      = mean;
        stats[64 + ch] = s2[0] * inv_n - mean * mean;
    }
}

__global__ __launch_bounds__(256) void bn_apply_kernel(float* __restrict__ x,
                                                       const float* __restrict__ stats,
                                                       const float* __restrict__ g,
                                                       const float* __restrict__ bta) {
    const int i  = blockIdx.x * 256 + threadIdx.x;
    const int ch = i & 63;
    const float mean = stats[ch], var = stats[64 + ch];
    x[i] = (x[i] - mean) * rsqrtf(var + 1e-5f) * g[ch] + bta[ch];
}

// ---------------------------------------------------------------------------
extern "C" void kernel_launch(void* const* d_in, const int* in_sizes, int n_in,
                              void* d_out, int out_size, void* d_ws, size_t ws_size,
                              hipStream_t stream) {
    const float* xyz   = (const float*)d_in[0];
    const float* feats = (const float*)d_in[1];
    const float* w_qs  = (const float*)d_in[2];
    const float* w_ks  = (const float*)d_in[3];
    const float* w_vs  = (const float*)d_in[4];
    const float* dw1   = (const float*)d_in[5];
    const float* db1   = (const float*)d_in[6];
    const float* dw2   = (const float*)d_in[7];
    const float* db2   = (const float*)d_in[8];
    const float* gw1   = (const float*)d_in[9];
    const float* gb1   = (const float*)d_in[10];
    const float* gw2   = (const float*)d_in[11];
    const float* gb2   = (const float*)d_in[12];
    const float* bng   = (const float*)d_in[13];
    const float* bnb   = (const float*)d_in[14];
    // d_in[15] = k (compile-time K_=16)

    char* ws = (char*)d_ws;
    int*       knn   = (int*)ws;                       // 2 MB
    float*     qbuf  = (float*)(ws + (2u << 20));      // 4 MB
    float*     kbuf  = (float*)(ws + (6u << 20));      // 4 MB
    float*     vbuf  = (float*)(ws + (10u << 20));     // 4 MB
    float*     stats = (float*)(ws + (14u << 20));     // 128 floats
    _Float16*  swz   = (_Float16*)(ws + (15u << 20));  // 56 KB pre-swizzled weights
    float*     out   = (float*)d_out;

    swz_weights_kernel<<<56, 32, 0, stream>>>(dw1, dw2, gw1, gw2, w_qs, w_ks, w_vs, swz);
    knn_kernel<<<dim3(N_ / KT, B_), KT, 0, stream>>>(xyz, knn);
    qkv_kernel<<<dim3((B_ * N_) / 64, 3), 128, 0, stream>>>(feats, swz, qbuf, kbuf, vbuf);
    attn_kernel<<<dim3((B_ * N_) / WV), 128, 0, stream>>>(xyz, feats, knn, swz,
                                                          qbuf, kbuf, vbuf,
                                                          db1, db2, gb1, gb2, out);
    bn_stats_kernel<<<64, 256, 0, stream>>>(out, stats);
    bn_apply_kernel<<<(B_ * N_ * D_) / 256, 256, 0, stream>>>(out, stats, bng, bnb);
}
